// PointTransformerCrossAttention_12386685681901
// MI455X (gfx1250) — matrix operand order repaired and checked by reading, not compiled, not run on hardware
//
#include <hip/hip_runtime.h>
#include <hip/hip_bf16.h>

typedef __attribute__((ext_vector_type(16))) __bf16 v16bf;
typedef __attribute__((ext_vector_type(8)))  __bf16 v8bf;
typedef __attribute__((ext_vector_type(8)))  float  v8f;

// ---- WMMA fragment layouts (wave32, ISA 7.12.2) ----------------------------
// A 16x32 bf16: lane L holds row M=L%16; elements = two contiguous runs:
//   e=0..7  -> K =  e      + 16*(L/16)   (k order used here, same as B)
//   e=8..15 -> K =  e      + 16*(L/16)
// B 32x16 bf16 (col-major source): lane L holds col N=L%16;
//   e=0..15 -> K = e + 16*(L/16)   (contiguous 32B when B stored [N][K])
// C/D 16x16 f32: element r -> row = r + 8*(L/16), col = L%16

__device__ __forceinline__ v16bf load_a_frag(const __bf16* __restrict__ rowp,
                                             int k0, int hf) {
    v8bf c0 = *(const v8bf*)(rowp + k0 + 16 * hf);
    v8bf c1 = *(const v8bf*)(rowp + k0 + 16 * hf + 8);
    v16bf a;
#pragma unroll
    for (int e = 0; e < 8; ++e) { a[e] = c0[e]; a[e + 8] = c1[e]; }
    return a;
}

// ---- elementwise converts ---------------------------------------------------
__global__ void f32_to_bf16_kernel(const float* __restrict__ in,
                                   __bf16* __restrict__ out, int n) {
    int i = blockIdx.x * 256 + threadIdx.x;
    if (i < n) out[i] = (__bf16)in[i];
}

// f32 row-major [R][C] -> bf16 col-major [C][R]
__global__ void f32_to_bf16_t_kernel(const float* __restrict__ in,
                                     __bf16* __restrict__ out, int R, int C) {
    int idx = blockIdx.x * 256 + threadIdx.x;
    if (idx >= R * C) return;
    int r = idx / C, c = idx % C;
    out[(long)c * R + r] = (__bf16)in[idx];
}

// bf16 row-major [R][C] -> bf16 col-major [C][R]
__global__ void transpose_bf16_kernel(const __bf16* __restrict__ in,
                                      __bf16* __restrict__ out, int R, int C) {
    int idx = blockIdx.x * 256 + threadIdx.x;
    if (idx >= R * C) return;
    int r = idx / C, c = idx % C;
    out[(long)c * R + r] = in[idx];
}

// Wp2 [256,8] f32 -> transposed+padded bf16 [16][256] (rows 8..15 zero)
__global__ void prep_wp2_kernel(const float* __restrict__ Wp2,
                                __bf16* __restrict__ out) {
    int i = blockIdx.x * 256 + threadIdx.x; // 0..4095
    int n = i >> 8, c = i & 255;
    out[i] = (__bf16)((n < 8) ? Wp2[c * 8 + n] : 0.0f);
}

// Aq[i,c] = xyz[i]·Wp1[:,c] (+ bp1[c] if addBias)   [N,256] f32
__global__ void pos_proj_kernel(const float* __restrict__ xyz,
                                const float* __restrict__ Wp1,
                                const float* __restrict__ bp1,
                                float* __restrict__ out, int N, int addBias) {
    int idx = blockIdx.x * 256 + threadIdx.x;
    if (idx >= N * 256) return;
    int i = idx >> 8, c = idx & 255;
    float v = addBias ? bp1[c] : 0.0f;
    v += xyz[i * 3 + 0] * Wp1[0 * 256 + c];
    v += xyz[i * 3 + 1] * Wp1[1 * 256 + c];
    v += xyz[i * 3 + 2] * Wp1[2 * 256 + c];
    out[idx] = v;
}

// ---- generic bf16 WMMA GEMM: C = A[MxK] * Bt[N][K] (+bias), batched via z --
// one wave per 16x16 C tile, 8 waves / block; K mult of 32, M,N mult of 16
template <typename CT>
__global__ void gemm_bf16_wmma(const __bf16* __restrict__ A, int lda, long aStride,
                               const __bf16* __restrict__ Bt, int ldb, long bStride,
                               const float* __restrict__ bias,
                               CT* __restrict__ C, int ldc, long cStride,
                               int M, int N, int K) {
    int b = blockIdx.z;
    A  += (long)b * aStride;
    Bt += (long)b * bStride;
    C  += (long)b * cStride;

    int wave = threadIdx.x >> 5;
    int lane = threadIdx.x & 31;
    int hf   = lane >> 4;
    int ln   = lane & 15;

    int tilesN = N >> 4;
    int tilesM = M >> 4;
    int tileId = blockIdx.x * 8 + wave;
    if (tileId >= tilesM * tilesN) return;
    int tm = tileId / tilesN, tn = tileId % tilesN;
    int row0 = tm * 16, col0 = tn * 16;

    const __bf16* arow_p = A + (long)(row0 + ln) * lda;
    const __bf16* bcol_p = Bt + (long)(col0 + ln) * ldb;

    v8f acc = {};
    for (int k0 = 0; k0 < K; k0 += 32) {
        v16bf af  = load_a_frag(arow_p, k0, hf);
        v16bf bfm = *(const v16bf*)(bcol_p + k0 + 16 * hf);
        acc = __builtin_amdgcn_wmma_f32_16x16x32_bf16(
            false, af, false, bfm, (short)0, acc, false, false);
    }
#pragma unroll
    for (int r = 0; r < 8; ++r) {
        int row = row0 + r + 8 * hf;
        float v = acc[r] + (bias ? bias[col0 + ln] : 0.0f);
        C[(long)row * ldc + (col0 + ln)] = (CT)v;
    }
}

// ---- fused score kernel: scores[h][i][j] = scale*q_h·k_h + rpe(i,j,h)+bp2 --
// block = 256 threads (8 waves); tile: 16 queries x 32 keys; all 8 heads.
__global__ void score_kernel(const float* __restrict__ Aq,   // [1024,256] (+bp1)
                             const float* __restrict__ Ak,   // [1024,256]
                             const __bf16* __restrict__ Qb,  // [1024,256]
                             const __bf16* __restrict__ Kb,  // [1024,256]
                             const __bf16* __restrict__ W2t, // [16][256] padded
                             const float* __restrict__ bp2,  // [8]
                             float* __restrict__ scores) {   // [8,1024,1024]
    __shared__ __align__(32) float  sAq[16 * 256];
    __shared__ __align__(32) float  sAk[32 * 256];
    __shared__ __align__(32) __bf16 sW2[16 * 256];
    __shared__ __align__(32) float  sRpe[512 * 8]; // [pair=il*32+jl][head]

    int i0 = blockIdx.x * 16;
    int j0 = blockIdx.y * 32;
    int t  = threadIdx.x;

    {   // vectorized staging
        const float4* srcQ = (const float4*)(Aq + (long)i0 * 256);
        const float4* srcK = (const float4*)(Ak + (long)j0 * 256);
        const uint4*  srcW = (const uint4*)W2t;
        float4* dQ = (float4*)sAq;
        float4* dK = (float4*)sAk;
        uint4*  dW = (uint4*)sW2;
        for (int idx = t; idx < 16 * 64; idx += 256) dQ[idx] = srcQ[idx];
        for (int idx = t; idx < 32 * 64; idx += 256) dK[idx] = srcK[idx];
        for (int idx = t; idx < 512; idx += 256)     dW[idx] = srcW[idx];
    }
    __syncthreads();

    int wave = t >> 5, lane = t & 31, hf = lane >> 4, ln = lane & 15;

    // RPE H-GEMM: H[512 pairs x 256] * Wp2[256 x 16]; 32 M-tiles, 4 per wave.
    // ks outer so the B fragment is loaded once per k-step (shared by 4 tiles).
    v8f acc0 = {}, acc1 = {}, acc2 = {}, acc3 = {};
    for (int ks = 0; ks < 8; ++ks) {
        int kc = ks * 32;
        v16bf bfm = *(const v16bf*)(&sW2[ln * 256 + kc + 16 * hf]);
        v8f* accs[4] = {&acc0, &acc1, &acc2, &acc3};
#pragma unroll
        for (int mti = 0; mti < 4; ++mti) {
            int mt = wave * 4 + mti;
            int il = mt >> 1;                 // pair = mt*16+ln -> row in Aq
            int jl = (mt & 1) * 16 + ln;      //                -> row in Ak
            v8f q0 = *(const v8f*)(&sAq[il * 256 + kc + 16 * hf]);
            v8f q1 = *(const v8f*)(&sAq[il * 256 + kc + 16 * hf + 8]);
            v8f k0v = *(const v8f*)(&sAk[jl * 256 + kc + 16 * hf]);
            v8f k1v = *(const v8f*)(&sAk[jl * 256 + kc + 16 * hf + 8]);
            v16bf af;
#pragma unroll
            for (int e = 0; e < 8; ++e) {
                af[e]     = (__bf16)fmaxf(q0[e] - k0v[e], 0.0f);
                af[e + 8] = (__bf16)fmaxf(q1[e] - k1v[e], 0.0f);
            }
            *accs[mti] = __builtin_amdgcn_wmma_f32_16x16x32_bf16(
                false, af, false, bfm, (short)0, *accs[mti], false, false);
        }
    }
    {
        v8f* accs[4] = {&acc0, &acc1, &acc2, &acc3};
#pragma unroll
        for (int mti = 0; mti < 4; ++mti) {
#pragma unroll
            for (int r = 0; r < 8; ++r) {
                int prow = (wave * 4 + mti) * 16 + r + 8 * hf;
                if (ln < 8) sRpe[prow * 8 + ln] = (*accs[mti])[r];
            }
        }
    }
    __syncthreads();

    // Energy: wave w handles head w; Dh=32 -> one WMMA per 16-key half
    int h = wave;
    const float scale = 0.17677669529663687f; // 1/sqrt(32)
    float bh = bp2[h];
    const __bf16* qrow = Qb + (long)(i0 + ln) * 256 + h * 32;
    v16bf afq = load_a_frag(qrow, 0, hf);
#pragma unroll
    for (int jh = 0; jh < 2; ++jh) {
        // K^T fragment: channel = e + 16*hf contiguous within a key's row
        v16bf bfm = *(const v16bf*)(Kb + (long)(j0 + jh * 16 + ln) * 256 +
                                    h * 32 + 16 * hf);
        v8f acc = {};
        acc = __builtin_amdgcn_wmma_f32_16x16x32_bf16(
            false, afq, false, bfm, (short)0, acc, false, false);
#pragma unroll
        for (int r = 0; r < 8; ++r) {
            int il = r + 8 * hf;
            int jl = jh * 16 + ln;
            float s = acc[r] * scale + sRpe[(il * 32 + jl) * 8 + h] + bh;
            scores[((long)h * 1024 + (i0 + il)) * 1024 + (j0 + jl)] = s;
        }
    }
}

// ---- row softmax over 1024, f32 scores -> bf16 probs -----------------------
__global__ void softmax_kernel(const float* __restrict__ scores,
                               __bf16* __restrict__ P) {
    long row = blockIdx.x; // h*1024 + i
    const float* s = scores + row * 1024;
    __bf16* p = P + row * 1024;
    __shared__ float red[8];
    __shared__ float bval[2];
    int t = threadIdx.x;

    float vals[4];
    float m = -1e30f;
#pragma unroll
    for (int k = 0; k < 4; ++k) { vals[k] = s[t + k * 256]; m = fmaxf(m, vals[k]); }
    for (int o = 16; o > 0; o >>= 1) m = fmaxf(m, __shfl_xor(m, o, 32));
    if ((t & 31) == 0) red[t >> 5] = m;
    __syncthreads();
    if (t == 0) {
        float mm = red[0];
        for (int w = 1; w < 8; ++w) mm = fmaxf(mm, red[w]);
        bval[0] = mm;
    }
    __syncthreads();
    m = bval[0];

    float lsum = 0.0f;
#pragma unroll
    for (int k = 0; k < 4; ++k) { vals[k] = __expf(vals[k] - m); lsum += vals[k]; }
    for (int o = 16; o > 0; o >>= 1) lsum += __shfl_xor(lsum, o, 32);
    if ((t & 31) == 0) red[t >> 5] = lsum;
    __syncthreads();
    if (t == 0) {
        float ss = 0.0f;
        for (int w = 0; w < 8; ++w) ss += red[w];
        bval[1] = ss;
    }
    __syncthreads();
    float inv = 1.0f / bval[1];
#pragma unroll
    for (int k = 0; k < 4; ++k) p[t + k * 256] = (__bf16)(vals[k] * inv);
}

// ---------------------------------------------------------------------------
extern "C" void kernel_launch(void* const* d_in, const int* in_sizes, int n_in,
                              void* d_out, int out_size, void* d_ws, size_t ws_size,
                              hipStream_t stream) {
    const float* q     = (const float*)d_in[0];
    const float* k     = (const float*)d_in[1];
    const float* v     = (const float*)d_in[2];
    const float* xyz_q = (const float*)d_in[3];
    const float* xyz_k = (const float*)d_in[4];
    const float* Wq    = (const float*)d_in[5];
    const float* bq    = (const float*)d_in[6];
    const float* Wk    = (const float*)d_in[7];
    const float* bk    = (const float*)d_in[8];
    const float* Wv    = (const float*)d_in[9];
    const float* bv    = (const float*)d_in[10];
    const float* Wo    = (const float*)d_in[11];
    const float* bo    = (const float*)d_in[12];
    const float* Wp1   = (const float*)d_in[13];
    const float* bp1   = (const float*)d_in[14];
    const float* Wp2   = (const float*)d_in[15];
    const float* bp2   = (const float*)d_in[16];
    float* out = (float*)d_out;

    const int N = 1024, D = 256;
    char* wp = (char*)d_ws;
    auto alloc = [&](size_t bytes) {
        char* p = wp;
        wp += (bytes + 255) & ~(size_t)255;
        return p;
    };
    __bf16* qb    = (__bf16*)alloc((size_t)N * D * 2);
    __bf16* kb    = (__bf16*)alloc((size_t)N * D * 2);
    __bf16* vb    = (__bf16*)alloc((size_t)N * D * 2);
    __bf16* WqT   = (__bf16*)alloc((size_t)D * D * 2);  // col-major [N][K]
    __bf16* WkT   = (__bf16*)alloc((size_t)D * D * 2);
    __bf16* WvT   = (__bf16*)alloc((size_t)D * D * 2);
    __bf16* WoT   = (__bf16*)alloc((size_t)D * D * 2);
    __bf16* W2t   = (__bf16*)alloc((size_t)16 * 256 * 2);
    float*  Aqf   = (float*)alloc((size_t)N * D * 4);
    float*  Akf   = (float*)alloc((size_t)N * D * 4);
    __bf16* Qb    = (__bf16*)alloc((size_t)N * D * 2);
    __bf16* Kb    = (__bf16*)alloc((size_t)N * D * 2);
    __bf16* Vb    = (__bf16*)alloc((size_t)N * D * 2);
    __bf16* VbT   = (__bf16*)alloc((size_t)D * N * 2);  // [256 ch][1024 keys]
    float*  Sc    = (float*)alloc((size_t)8 * N * N * 4);   // 32 MB
    __bf16* Pb    = (__bf16*)alloc((size_t)8 * N * N * 2);  // 16 MB
    __bf16* AO    = (__bf16*)alloc((size_t)N * D * 2);

    // converts / layout prep
    f32_to_bf16_kernel<<<(N * D + 255) / 256, 256, 0, stream>>>(q, qb, N * D);
    f32_to_bf16_kernel<<<(N * D + 255) / 256, 256, 0, stream>>>(k, kb, N * D);
    f32_to_bf16_kernel<<<(N * D + 255) / 256, 256, 0, stream>>>(v, vb, N * D);
    f32_to_bf16_t_kernel<<<(D * D + 255) / 256, 256, 0, stream>>>(Wq, WqT, D, D);
    f32_to_bf16_t_kernel<<<(D * D + 255) / 256, 256, 0, stream>>>(Wk, WkT, D, D);
    f32_to_bf16_t_kernel<<<(D * D + 255) / 256, 256, 0, stream>>>(Wv, WvT, D, D);
    f32_to_bf16_t_kernel<<<(D * D + 255) / 256, 256, 0, stream>>>(Wo, WoT, D, D);
    prep_wp2_kernel<<<16, 256, 0, stream>>>(Wp2, W2t);
    pos_proj_kernel<<<(N * D + 255) / 256, 256, 0, stream>>>(xyz_q, Wp1, bp1, Aqf, N, 1);
    pos_proj_kernel<<<(N * D + 255) / 256, 256, 0, stream>>>(xyz_k, Wp1, bp1, Akf, N, 0);

    // QKV projections: M=1024,N=256,K=256 -> 1024 tiles / 8 waves = 128 blocks
    gemm_bf16_wmma<__bf16><<<dim3(128, 1, 1), 256, 0, stream>>>(
        qb, D, 0, WqT, D, 0, bq, Qb, D, 0, N, D, D);
    gemm_bf16_wmma<__bf16><<<dim3(128, 1, 1), 256, 0, stream>>>(
        kb, D, 0, WkT, D, 0, bk, Kb, D, 0, N, D, D);
    gemm_bf16_wmma<__bf16><<<dim3(128, 1, 1), 256, 0, stream>>>(
        vb, D, 0, WvT, D, 0, bv, Vb, D, 0, N, D, D);

    // V -> col-major for the P@V GEMM
    transpose_bf16_kernel<<<(N * D + 255) / 256, 256, 0, stream>>>(Vb, VbT, N, D);

    // fused scores (RPE MLP via WMMA + QK^T via WMMA)
    score_kernel<<<dim3(64, 32, 1), 256, 0, stream>>>(Aqf, Akf, Qb, Kb, W2t, bp2, Sc);

    // softmax per (head,row)
    softmax_kernel<<<8 * N, 256, 0, stream>>>(Sc, Pb);

    // P @ V per head: M=1024,N=32,K=1024; 128 tiles -> 16 blocks, z=8 heads
    // B = VbT + 32h rows of [1024]; col n -> VbT[(32h+n)*1024 + k]
    gemm_bf16_wmma<__bf16><<<dim3(16, 1, 8), 256, 0, stream>>>(
        Pb, N, (long)N * N, VbT, N, (long)32 * N, nullptr, AO, D, 32, N, 32, N);

    // output projection -> f32 out
    gemm_bf16_wmma<float><<<dim3(128, 1, 1), 256, 0, stream>>>(
        AO, D, 0, WoT, D, 0, bo, out, D, 0, N, D, D);

    (void)in_sizes; (void)n_in; (void)out_size; (void)ws_size;
}
